// PirouetteAgent_63677185131364
// MI455X (gfx1250) — matrix operand order repaired and checked by reading, not compiled, not run on hardware
//
#include <hip/hip_runtime.h>
#include <cmath>

// ---------------- CDNA5 WMMA types ----------------
typedef __attribute__((ext_vector_type(16))) __bf16 v16bf;
typedef __attribute__((ext_vector_type(8)))  float  v8f;

union BF16Frag { v16bf v; uint4 q[2]; };

__device__ __forceinline__ unsigned short f2bf_bits(float f) {
  return __builtin_bit_cast(unsigned short, (__bf16)f);
}

// ---------------------------------------------------------------------------
// Prep kernel: repack a row-major fp32 weight [K x ldw] (using columns
// [0,Nsrc)) into bf16 WMMA-B fragment-major layout:
//   out[((nt*(K/32)+kc)*32 + lane)*16 + e] = W[kc*32 + (e%8) + 8*(lane/16) + 16*(e/8)][nt*16 + lane%16]
// so the main kernel fetches one whole v16bf B-fragment with a single
// contiguous 32-byte load per lane.
// ---------------------------------------------------------------------------
__global__ void pack_b_kernel(const float* __restrict__ W, unsigned short* __restrict__ out,
                              int K, int Nsrc, int ldw, int Npack)
{
  int idx = blockIdx.x * 256 + threadIdx.x;
  int total = (Npack >> 4) * (K >> 5) * 512;
  if (idx >= total) return;
  int e    = idx & 15;
  int lane = (idx >> 4) & 31;
  int tile = idx >> 9;
  int kcn  = K >> 5;
  int kc   = tile % kcn;
  int nt   = tile / kcn;
  int k = kc * 32 + (e & 7) + 8 * (lane >> 4) + 16 * (e >> 3);
  int n = nt * 16 + (lane & 15);
  float v = (n < Nsrc) ? W[(size_t)k * ldw + n] : 0.f;
  out[idx] = f2bf_bits(v);
}

// ---------------------------------------------------------------------------
// Prep kernel: fold the real-FFT cosine matrix into Wk1 and pack.
//   F[j][o] = sum_t cos(2*pi*t*j/256) * Wk1[t][o]      (K=256 in, N=128 out)
// ---------------------------------------------------------------------------
__global__ void pack_wk1_kernel(const float* __restrict__ Wk1, unsigned short* __restrict__ out)
{
  __shared__ float ct[256];
  int tid = threadIdx.x;
  ct[tid] = cosf(0.02454369260617026f * (float)tid);   // 2*pi/256
  __syncthreads();

  int idx = blockIdx.x * 256 + tid;                    // total = 64*512 = 32768
  int e = idx & 15, lane = (idx >> 4) & 31, tile = idx >> 9;
  int kc = tile & 7, nt = tile >> 3;
  int j = kc * 32 + (e & 7) + 8 * (lane >> 4) + 16 * (e >> 3);              // input feature index
  int o = nt * 16 + (lane & 15);                       // output column
  float acc = 0.f;
  for (int t = 0; t < 256; ++t)
    acc += ct[(t * j) & 255] * Wk1[t * 128 + o];
  out[idx] = f2bf_bits(acc);
}

// Paired relu + bf16 conversion + LDS writeback of one v8f D-tile column.
// Packs element pairs so the compiler emits one v_cvt_pk_bf16_f32 per pair.
__device__ __forceinline__ void writeback_tile(__bf16* __restrict__ L, int ld, int n,
                                               int half, v8f acc, bool relu)
{
#pragma unroll
  for (int r = 0; r < 8; r += 2) {
    float v0 = acc[r], v1 = acc[r + 1];
    if (relu) { v0 = fmaxf(v0, 0.f); v1 = fmaxf(v1, 0.f); }
    union { __bf16 h[2]; unsigned u; } pk;
    pk.h[0] = (__bf16)v0;
    pk.h[1] = (__bf16)v1;
    L[(r + 0 + 8 * half) * ld + n] = pk.h[0];
    L[(r + 1 + 8 * half) * ld + n] = pk.h[1];   // -> ds_store_b16_d16_hi
  }
}

// ---------------------------------------------------------------------------
// NT independent 16x16 output tiles sharing one A stream (16xK bf16 in LDS,
// row-major, row stride = K).  Software-pipelined: A + all NT B fragments for
// chunk kc+1 are loaded into distinct live registers before the NT WMMAs of
// chunk kc issue, so vmem latency overlaps the WMMA group instead of
// serializing (load -> s_wait 0 -> wmma).
// ---------------------------------------------------------------------------
template<int K, int NT>
__device__ __forceinline__ void wmma_rows(const __bf16* __restrict__ A,
                                          const unsigned short* __restrict__ Bp,
                                          int nt0, const float* __restrict__ bias,
                                          __bf16* __restrict__ L, int ld, bool relu)
{
  constexpr int KC = K / 32;
  const int lane = threadIdx.x & 31;
  const int half = lane >> 4;
  const int nloc = lane & 15;
  const __bf16* arow = A + nloc * K;

  v8f acc[NT];
#pragma unroll
  for (int c = 0; c < NT; ++c) {
    float bv = bias[(nt0 + c) * 16 + nloc];
#pragma unroll
    for (int i = 0; i < 8; ++i) acc[c][i] = bv;
  }

  // Per-chain B fragment streams (chain c, chunk kc at bq[c] + 2*kc).
  const uint4* bq[NT];
#pragma unroll
  for (int c = 0; c < NT; ++c)
    bq[c] = (const uint4*)(Bp + ((size_t)(nt0 + c) * KC * 32 + lane) * 16);

  BF16Frag a_cur;
  BF16Frag b_cur[NT];
  {
    const uint4* ap = (const uint4*)(arow + 8 * half);
    a_cur.q[0] = ap[0];
    a_cur.q[1] = ap[2];
#pragma unroll
    for (int c = 0; c < NT; ++c) { b_cur[c].q[0] = bq[c][0]; b_cur[c].q[1] = bq[c][1]; }
  }

#pragma unroll
  for (int kc = 0; kc < KC; ++kc) {
    BF16Frag a_nxt;
    BF16Frag b_nxt[NT];
    if (kc + 1 < KC) {
      const uint4* ap = (const uint4*)(arow + (kc + 1) * 32 + 8 * half);
      a_nxt.q[0] = ap[0];
      a_nxt.q[1] = ap[2];
#pragma unroll
      for (int c = 0; c < NT; ++c) {
        const uint4* p = bq[c] + (size_t)(kc + 1) * 2;
        b_nxt[c].q[0] = p[0];
        b_nxt[c].q[1] = p[1];
      }
    }
#pragma unroll
    for (int c = 0; c < NT; ++c)
      acc[c] = __builtin_amdgcn_wmma_f32_16x16x32_bf16(false, a_cur.v, false, b_cur[c].v,
                                                       (short)0, acc[c], false, false);
    if (kc + 1 < KC) {
      a_cur = a_nxt;
#pragma unroll
      for (int c = 0; c < NT; ++c) b_cur[c] = b_nxt[c];
    }
  }

#pragma unroll
  for (int c = 0; c < NT; ++c)
    writeback_tile(L, ld, (nt0 + c) * 16 + nloc, half, acc[c], relu);
}

// Single 16x16 output tile: two split-K accumulator chains, software-pipelined.
template<int K>
__device__ __forceinline__ v8f wmma_one(const __bf16* __restrict__ A,
                                        const unsigned short* __restrict__ Bp,
                                        float biasv)
{
  constexpr int T = K / 64;                 // pair steps
  static_assert(K % 64 == 0, "split-K needs even chunk count");
  const int lane = threadIdx.x & 31;
  const int half = lane >> 4;
  const __bf16* arow = A + (lane & 15) * K;
  const uint4* bq = (const uint4*)(Bp + (size_t)lane * 16);

  v8f acc0, acc1;
#pragma unroll
  for (int i = 0; i < 8; ++i) { acc0[i] = biasv; acc1[i] = 0.f; }

  BF16Frag a0c, a1c, b0c, b1c;
  {
    const uint4* ap0 = (const uint4*)(arow + 8 * half);
    a0c.q[0] = ap0[0]; a0c.q[1] = ap0[2];
    const uint4* ap1 = (const uint4*)(arow + 32 + 8 * half);
    a1c.q[0] = ap1[0]; a1c.q[1] = ap1[2];
    b0c.q[0] = bq[0]; b0c.q[1] = bq[1];
    b1c.q[0] = bq[2]; b1c.q[1] = bq[3];
  }

#pragma unroll
  for (int t = 0; t < T; ++t) {
    BF16Frag a0n, a1n, b0n, b1n;
    if (t + 1 < T) {
      const int kc = (t + 1) * 2;
      const uint4* ap0 = (const uint4*)(arow + kc * 32 + 8 * half);
      a0n.q[0] = ap0[0]; a0n.q[1] = ap0[2];
      const uint4* ap1 = (const uint4*)(arow + (kc + 1) * 32 + 8 * half);
      a1n.q[0] = ap1[0]; a1n.q[1] = ap1[2];
      const uint4* p = bq + (size_t)kc * 2;
      b0n.q[0] = p[0]; b0n.q[1] = p[1];
      b1n.q[0] = p[2]; b1n.q[1] = p[3];
    }
    acc0 = __builtin_amdgcn_wmma_f32_16x16x32_bf16(false, a0c.v, false, b0c.v,
                                                   (short)0, acc0, false, false);
    acc1 = __builtin_amdgcn_wmma_f32_16x16x32_bf16(false, a1c.v, false, b1c.v,
                                                   (short)0, acc1, false, false);
    if (t + 1 < T) { a0c = a0n; a1c = a1n; b0c = b0n; b1c = b1n; }
  }
  return acc0 + acc1;
}

// ---------------------------------------------------------------------------
// Fused forward kernel: 8 wave32 per block, each wave owns 16 rows and a
// private LDS slice.  All matmuls are v_wmma_f32_16x16x32_bf16.
// ---------------------------------------------------------------------------
#define WAVES 8

__global__ __launch_bounds__(256) void fused_kernel(
    const float* __restrict__ state,
    const float* __restrict__ b1,  const float* __restrict__ b2,
    const float* __restrict__ bid, const float* __restrict__ bk1,
    const float* __restrict__ bk2, const float* __restrict__ ba1,
    const float* __restrict__ ba2,
    const unsigned short* __restrict__ W1p,  const unsigned short* __restrict__ W2p,
    const unsigned short* __restrict__ Widp, const unsigned short* __restrict__ Wk1p,
    const unsigned short* __restrict__ Wk2p, const unsigned short* __restrict__ Wa1p,
    const unsigned short* __restrict__ Wa2p,
    float* __restrict__ out, int Nrows)
{
  __shared__ __bf16 sb1[WAVES][16 * 256];
  __shared__ __bf16 sb2[WAVES][16 * 256];
  __shared__ float  sid[WAVES][16][16];
  __shared__ int    spred[WAVES][16];

  const int tid  = threadIdx.x;
  const int wave = tid >> 5, lane = tid & 31;
  const int half = lane >> 4, nloc = lane & 15;
  const int rowbase = (blockIdx.x * WAVES + wave) * 16;

  __bf16* buf1 = sb1[wave];
  __bf16* buf2 = sb2[wave];

  // ---- stage state tile (16 x 64) as bf16 into buf1 (row stride 64) ----
  {
    const int m = nloc, c0 = half * 32;
    int srow = rowbase + m;
    if (srow >= Nrows) srow = Nrows - 1;
    const float4* src = (const float4*)(state + (size_t)srow * 64 + c0);
#pragma unroll
    for (int j = 0; j < 8; ++j) {
      float4 f = src[j];
      union { __bf16 h[4]; uint2 u; } p;
      p.h[0] = (__bf16)f.x; p.h[1] = (__bf16)f.y;
      p.h[2] = (__bf16)f.z; p.h[3] = (__bf16)f.w;
      *(uint2*)(buf1 + m * 64 + c0 + j * 4) = p.u;
    }
  }
  __syncthreads();

  // ---- GEMM1: feat = relu(state @ W1 + b1) -> buf2 [16][256] ----
#pragma unroll
  for (int nt = 0; nt < 16; nt += 4)
    wmma_rows<64, 4>(buf1, W1p, nt, b1, buf2, 256, true);
  __syncthreads();

  // ---- GEMM2: feat2 = relu(feat @ W2 + b2) -> buf1 [16][256] ----
#pragma unroll
  for (int nt = 0; nt < 16; nt += 4)
    wmma_rows<256, 4>(buf2, W2p, nt, b2, buf1, 256, true);
  __syncthreads();

  // ---- id head (Wid padded to 16 cols) + per-row argmax over 4 logits ----
  {
    float bb = (nloc < 4) ? bid[nloc] : 0.f;
    v8f acc = wmma_one<256>(buf1, Widp, bb);
#pragma unroll
    for (int r = 0; r < 8; ++r) sid[wave][r + 8 * half][nloc] = acc[r];
  }
  __syncthreads();
  if (lane < 16) {
    const float* row = sid[wave][lane];
    int best = 0; float bv = row[0];
#pragma unroll
    for (int j = 1; j < 4; ++j) { if (row[j] > bv) { bv = row[j]; best = j; } }
    spred[wave][lane] = best;
  }
  __syncthreads();

  // ---- corr1: hcorr = relu(feat2 @ (C@Wk1) + bk1) -> buf2 [16][128] ----
#pragma unroll
  for (int nt = 0; nt < 8; nt += 4)
    wmma_rows<256, 4>(buf1, Wk1p, nt, bk1, buf2, 128, true);
  __syncthreads();

  // ---- corr2: corr = hcorr @ Wk2 + bk2  (16x16 tile kept in registers) ----
  v8f corr = wmma_one<128>(buf2, Wk2p, bk2[nloc]);
  __syncthreads();

  int pr[8];
#pragma unroll
  for (int r = 0; r < 8; ++r) pr[r] = spred[wave][r + 8 * half];

  v8f sel;
#pragma unroll
  for (int r = 0; r < 8; ++r) sel[r] = 0.f;

  // ---- experts: h_e = relu(feat2 @ Wa1[e] + ba1[e]); out_e = h_e @ Wa2[e][:,:16] + ba2[e][:16]
  for (int e = 0; e < 4; ++e) {
    const unsigned short* wa1 = Wa1p + (size_t)e * 65536;
#pragma unroll
    for (int nt = 0; nt < 16; nt += 4)
      wmma_rows<256, 4>(buf1, wa1, nt, ba1 + e * 256, buf2, 256, true);
    __syncthreads();
    v8f oe = wmma_one<256>(buf2, Wa2p + (size_t)e * 4096, ba2[e * 32 + nloc]);
#pragma unroll
    for (int r = 0; r < 8; ++r) if (pr[r] == e) sel[r] = oe[r];
    __syncthreads();
  }

  // ---- final: tanh(selected mean + corr) -> out [N][16] ----
#pragma unroll
  for (int r = 0; r < 8; ++r) {
    int row = rowbase + r + 8 * half;
    if (row < Nrows) out[(size_t)row * 16 + nloc] = tanhf(sel[r] + corr[r]);
  }
}

// ---------------------------------------------------------------------------
extern "C" void kernel_launch(void* const* d_in, const int* in_sizes, int n_in,
                              void* d_out, int out_size, void* d_ws, size_t ws_size,
                              hipStream_t stream)
{
  const float* state = (const float*)d_in[0];
  const float* W1  = (const float*)d_in[1];
  const float* b1  = (const float*)d_in[2];
  const float* W2  = (const float*)d_in[3];
  const float* b2  = (const float*)d_in[4];
  const float* Wid = (const float*)d_in[5];
  const float* bid = (const float*)d_in[6];
  const float* Wk1 = (const float*)d_in[7];
  const float* bk1 = (const float*)d_in[8];
  const float* Wk2 = (const float*)d_in[9];
  const float* bk2 = (const float*)d_in[10];
  const float* Wa1 = (const float*)d_in[11];
  const float* ba1 = (const float*)d_in[12];
  const float* Wa2 = (const float*)d_in[13];
  const float* ba2 = (const float*)d_in[14];
  float* out = (float*)d_out;
  const int Nrows = in_sizes[0] / 64;

  // bf16 packed-weight layout inside workspace (ushort units)
  unsigned short* ws   = (unsigned short*)d_ws;
  unsigned short* W1p  = ws;              // 16384
  unsigned short* W2p  = W1p  + 16384;    // 65536
  unsigned short* Widp = W2p  + 65536;    // 4096
  unsigned short* Wk1p = Widp + 4096;     // 32768
  unsigned short* Wk2p = Wk1p + 32768;    // 2048
  unsigned short* Wa1p = Wk2p + 2048;     // 262144
  unsigned short* Wa2p = Wa1p + 262144;   // 16384

  auto nb = [](int t) { return (t + 255) / 256; };
  pack_b_kernel<<<nb(16384), 256, 0, stream>>>(W1,  W1p,  64,  256, 256, 256);
  pack_b_kernel<<<nb(65536), 256, 0, stream>>>(W2,  W2p,  256, 256, 256, 256);
  pack_b_kernel<<<nb(4096),  256, 0, stream>>>(Wid, Widp, 256, 4,   4,   16);
  pack_wk1_kernel<<<nb(32768), 256, 0, stream>>>(Wk1, Wk1p);
  pack_b_kernel<<<nb(2048),  256, 0, stream>>>(Wk2, Wk2p, 128, 16,  16,  16);
  for (int e = 0; e < 4; ++e) {
    pack_b_kernel<<<nb(65536), 256, 0, stream>>>(Wa1 + (size_t)e * 65536,
                                                 Wa1p + (size_t)e * 65536,
                                                 256, 256, 256, 256);
    pack_b_kernel<<<nb(4096), 256, 0, stream>>>(Wa2 + (size_t)e * 8192,
                                                Wa2p + (size_t)e * 4096,
                                                256, 16, 32, 16);
  }

  const int blocks = (Nrows + 16 * WAVES - 1) / (16 * WAVES);
  fused_kernel<<<blocks, 256, 0, stream>>>(state, b1, b2, bid, bk1, bk2, ba1, ba2,
                                           W1p, W2p, Widp, Wk1p, Wk2p, Wa1p, Wa2p,
                                           out, Nrows);
  (void)n_in; (void)out_size; (void)ws_size;
}
